// DFlashAttention_73632919322983
// MI455X (gfx1250) — compile-verified, matrix operand order checked by reading, not run on hardware
//
#include <hip/hip_runtime.h>
#include <hip/hip_bf16.h>

// ---------------- problem constants ----------------
#define DDIM    2048
#define NHEADS  16
#define KHEADS  8
#define HDIM    128
#define HHALF   64
#define TNOISE  1024
#define TPAD    1024
#define TNEW    2048
#define MAXKV   4096
#define SMSCALE 0.08838834764831845f   // 128^-0.5
#define RMSEPS  1e-6f
#define NEGINF  (-1e30f)
#define LOG_THETA 13.815510557964274f  // ln(1e6)

typedef __attribute__((ext_vector_type(16))) __bf16 v16bf;
typedef __attribute__((ext_vector_type(8)))  float  v8f;

__device__ __forceinline__ v8f wmma_bf16(v16bf a, v16bf b, v8f c) {
  return __builtin_amdgcn_wmma_f32_16x16x32_bf16(false, a, false, b, (short)0, c,
                                                 false, false);
}

// K index for bf16-pair v of a 16x32 A/B fragment (ISA 7.12.2 16-bit layout):
// VGPR 0..3 -> K 0..7 (lanes 0-15) / 8..15 (lanes 16-31); VGPR 4..7 -> +16.
__device__ __forceinline__ int frag_k0(int v, int lane) {
  return ((v & 4) << 2) + (((lane >> 4) & 1) << 3) + ((v & 3) << 1);
}

__device__ __forceinline__ v16bf load_a_lds_bf16(const __bf16* t, int stride, int lane) {
  const int m = lane & 15;
  v16bf a;
#pragma unroll
  for (int v = 0; v < 8; ++v) {
    const int k0 = frag_k0(v, lane);
    a[2 * v]     = t[m * stride + k0];
    a[2 * v + 1] = t[m * stride + k0 + 1];
  }
  return a;
}

__device__ __forceinline__ v16bf load_a_lds_f32(const float* t, int stride, int lane) {
  const int m = lane & 15;
  v16bf a;
#pragma unroll
  for (int v = 0; v < 8; ++v) {
    const int k0 = frag_k0(v, lane);
    a[2 * v]     = (__bf16)t[m * stride + k0];
    a[2 * v + 1] = (__bf16)t[m * stride + k0 + 1];
  }
  return a;
}

// B fragment (32xK x 16N): element (k, n) at base[k*strideK + n*strideN], f32->bf16.
__device__ __forceinline__ v16bf load_b_f32(const float* base, int strideK, int strideN,
                                            int lane) {
  const int n = lane & 15;
  v16bf b;
#pragma unroll
  for (int v = 0; v < 8; ++v) {
    const int k0 = frag_k0(v, lane);
    b[2 * v]     = (__bf16)base[(size_t)k0 * strideK + (size_t)n * strideN];
    b[2 * v + 1] = (__bf16)base[(size_t)(k0 + 1) * strideK + (size_t)n * strideN];
  }
  return b;
}

// Scatter one k/v row-element into the output cache, reproducing the reference's
// dynamic_update_slice ordering (ctx slice first, noise slice overwrites its tail).
__device__ __forceinline__ void scatter_kv(float* cache, int khead, int t, int h,
                                           float val, int cache_len, int actx) {
  long p;
  if (t < TPAD) {
    if (t >= actx) {
      if (t - actx < TNOISE) return;  // this slot is overwritten by a noise row
      val = 0.0f;                     // masked ctx row that survives
    }
    p = (long)cache_len + t;
  } else {
    p = (long)cache_len + actx + (t - TPAD);
  }
  // GQA repeat G=2: cache head n uses source head n/2
  const size_t o0 = (((size_t)(2 * khead) * MAXKV) + (size_t)p) * HDIM + h;
  const size_t o1 = (((size_t)(2 * khead + 1) * MAXKV) + (size_t)p) * HDIM + h;
  cache[o0] = val;
  cache[o1] = val;
}

// ---------------- kernel 1: Q/K/V projection + RMSNorm + RoPE + cache scatter ----
// grid: (128 rowtiles, 16 heads, 3={Q,K,V}), block 128 (4 waves).
// Block computes a 16-token x 128-col (one head) tile with bf16 WMMA.
__global__ __launch_bounds__(128) void proj_kernel(
    const float* __restrict__ x_noise, const float* __restrict__ target_hidden,
    const int* __restrict__ noise_pos, const int* __restrict__ ctx_pos,
    const float* __restrict__ Wq, const float* __restrict__ Wk,
    const float* __restrict__ Wv, const float* __restrict__ q_scale,
    const float* __restrict__ k_scale, const int* __restrict__ p_cache_len,
    const int* __restrict__ p_actx, float* __restrict__ q_ws,
    float* __restrict__ kcache, float* __restrict__ vcache) {
  const int which = blockIdx.z;  // 0=Q, 1=K, 2=V
  const int rt = blockIdx.x;
  const int head = blockIdx.y;
  if (which == 0) {
    if (rt >= TNOISE / 16 || head >= NHEADS) return;
  } else {
    if (head >= KHEADS) return;  // rt < 128 covers TNEW rows
  }

  const int lane = threadIdx.x & 31;
  const int wave = threadIdx.x >> 5;
  const int r0 = rt * 16;

  const float* W = (which == 0) ? Wq : (which == 1) ? Wk : Wv;
  const int ldW = ((which == 0) ? NHEADS : KHEADS) * HDIM;
  const int col0 = head * HDIM + wave * 32;

  __shared__ __bf16 As[16 * 32];
  __shared__ float Ys[16 * HDIM];
  __shared__ float rstd[16];

  v8f acc0 = {};
  v8f acc1 = {};

  for (int kk = 0; kk < DDIM; kk += 32) {
    __syncthreads();
    for (int i = threadIdx.x; i < 512; i += 128) {
      const int mr = i >> 5, kc = i & 31;
      const int row = r0 + mr;
      const float* src;
      if (which == 0)
        src = x_noise + (size_t)row * DDIM;
      else
        src = (row < TPAD) ? target_hidden + (size_t)row * DDIM
                           : x_noise + (size_t)(row - TPAD) * DDIM;
      As[i] = (__bf16)src[kk + kc];
    }
    __syncthreads();
    v16bf a = load_a_lds_bf16(As, 32, lane);
    v16bf b0 = load_b_f32(W + (size_t)kk * ldW + col0, ldW, 1, lane);
    v16bf b1 = load_b_f32(W + (size_t)kk * ldW + col0 + 16, ldW, 1, lane);
    acc0 = wmma_bf16(a, b0, acc0);
    acc1 = wmma_bf16(a, b1, acc1);
  }

  __syncthreads();
  {
    const int n = lane & 15, mb = (lane >> 4) << 3;
#pragma unroll
    for (int r = 0; r < 8; ++r) {
      Ys[(mb + r) * HDIM + wave * 32 + n] = acc0[r];
      Ys[(mb + r) * HDIM + wave * 32 + 16 + n] = acc1[r];
    }
  }
  __syncthreads();

  const int cache_len = *p_cache_len;
  const int actx = *p_actx;
  const int m = threadIdx.x >> 3;          // 16 tokens, 8 threads each
  const int hbase = (threadIdx.x & 7) << 4;  // 16 h-values per thread
  const int t = r0 + m;

  if (which == 2) {  // V: no norm/rope, just ctx-mask + scatter
#pragma unroll
    for (int i = 0; i < 16; ++i) {
      const int h = hbase + i;
      scatter_kv(vcache, head, t, h, Ys[m * HDIM + h], cache_len, actx);
    }
    return;
  }

  if (threadIdx.x < 16) {
    float s = 0.f;
    for (int h = 0; h < HDIM; ++h) {
      const float y = Ys[threadIdx.x * HDIM + h];
      s += y * y;
    }
    rstd[threadIdx.x] = rsqrtf(s * (1.0f / HDIM) + RMSEPS);
  }
  __syncthreads();

  const float rs = rstd[m];
  const float* sc = (which == 0) ? q_scale : k_scale;
  const int pos =
      (which == 0) ? noise_pos[t] : (t < TPAD ? ctx_pos[t] : noise_pos[t - TPAD]);
#pragma unroll
  for (int i = 0; i < 16; ++i) {
    const int h = hbase + i;
    const int fi = h & (HHALF - 1);
    const float invf = __expf(-(float)fi * (LOG_THETA / (float)HHALF));
    const float ang = (float)pos * invf;
    float sv, cv;
    sincosf(ang, &sv, &cv);
    const float x1 = Ys[m * HDIM + fi] * rs * sc[fi];
    const float x2 = Ys[m * HDIM + fi + HHALF] * rs * sc[fi + HHALF];
    const float val = (h < HHALF) ? (x1 * cv - x2 * sv) : (x2 * cv + x1 * sv);
    if (which == 0)
      q_ws[((size_t)t * NHEADS + head) * HDIM + h] = val;
    else
      scatter_kv(kcache, head, t, h, val, cache_len, actx);
  }
}

// ---------------- kernel 2: flash attention (online softmax) --------------------
// grid: (16 heads, 64 q-tiles), block 128 (4 waves). Each wave owns 32 of the 128
// output columns; each wave computes one 16x16 score tile per 64-key chunk.
__global__ __launch_bounds__(128) void attn_kernel(
    const float* __restrict__ qbuf, const float* __restrict__ kcache,
    const float* __restrict__ vcache, const int* __restrict__ p_cache_len,
    const int* __restrict__ p_actx, float* __restrict__ attn_ws) {
  const int head = blockIdx.x;
  const int qt = blockIdx.y;
  const int lane = threadIdx.x & 31;
  const int wave = threadIdx.x >> 5;
  const int new_len = *p_cache_len + *p_actx + TNOISE;

  __shared__ __bf16 Qs[16 * HDIM];
  __shared__ float Ss[16 * 64];
  __shared__ float Mrow[16], Lrow[16], Arow[16];

  for (int i = threadIdx.x; i < 16 * HDIM; i += 128) {
    const int mm = i >> 7, h = i & (HDIM - 1);
    Qs[i] = (__bf16)(qbuf[((size_t)(qt * 16 + mm) * NHEADS + head) * HDIM + h] *
                     SMSCALE);  // fold softmax scale into q
  }
  if (threadIdx.x < 16) {
    Mrow[threadIdx.x] = -3.0e38f;
    Lrow[threadIdx.x] = 0.f;
  }
  __syncthreads();

  const float* kb_head = kcache + (size_t)head * MAXKV * HDIM;
  const float* vb_head = vcache + (size_t)head * MAXKV * HDIM;

  v8f o0 = {};
  v8f o1 = {};

  for (int p0 = 0; p0 < new_len; p0 += 64) {
    const int kp = p0 + wave * 16;
    v8f s = {};
    const float* kb = kb_head + (size_t)kp * HDIM;
#pragma unroll
    for (int hh = 0; hh < HDIM; hh += 32) {
      v16bf a = load_a_lds_bf16(Qs + hh, HDIM, lane);
      v16bf b = load_b_f32(kb + hh, 1, HDIM, lane);  // K^T: (h, j) -> kb[j*H + h]
      s = wmma_bf16(a, b, s);
    }
    {
      const int n = lane & 15, mb = (lane >> 4) << 3;
#pragma unroll
      for (int r = 0; r < 8; ++r) {
        float sv = s[r];
        if (kp + n >= new_len) sv = NEGINF;  // kv_valid mask
        Ss[(mb + r) * 64 + wave * 16 + n] = sv;
      }
    }
    __syncthreads();
    if (threadIdx.x < 16) {
      const int mm = threadIdx.x;
      float mx = Mrow[mm];
      for (int j = 0; j < 64; ++j) mx = fmaxf(mx, Ss[mm * 64 + j]);
      const float al = __expf(Mrow[mm] - mx);
      float l = Lrow[mm] * al;
      for (int j = 0; j < 64; ++j) {
        const float pr = __expf(Ss[mm * 64 + j] - mx);
        Ss[mm * 64 + j] = pr;
        l += pr;
      }
      Mrow[mm] = mx;
      Lrow[mm] = l;
      Arow[mm] = al;
    }
    __syncthreads();
    {
      const int mb = (lane >> 4) << 3;
#pragma unroll
      for (int r = 0; r < 8; ++r) {
        const float al = Arow[mb + r];
        o0[r] *= al;
        o1[r] *= al;
      }
    }
    const float* vb = vb_head + (size_t)p0 * HDIM + wave * 32;
#pragma unroll
    for (int kc = 0; kc < 64; kc += 32) {
      v16bf pa = load_a_lds_f32(Ss + kc, 64, lane);
      v16bf b0 = load_b_f32(vb + (size_t)kc * HDIM, HDIM, 1, lane);
      v16bf b1 = load_b_f32(vb + (size_t)kc * HDIM + 16, HDIM, 1, lane);
      o0 = wmma_bf16(pa, b0, o0);
      o1 = wmma_bf16(pa, b1, o1);
    }
    __syncthreads();
  }

  {
    const int n = lane & 15, mb = (lane >> 4) << 3;
#pragma unroll
    for (int r = 0; r < 8; ++r) {
      const int mm = mb + r;
      const float inv = 1.0f / Lrow[mm];
      const size_t base =
          ((size_t)(qt * 16 + mm) * NHEADS + head) * HDIM + wave * 32 + n;
      attn_ws[base] = o0[r] * inv;
      attn_ws[base + 16] = o1[r] * inv;
    }
  }
}

// ---------------- kernel 3: output projection out = attn @ Wo -------------------
// attn flat [1024 x 2048], Wo flat [(n*H)=2048 x 2048]. grid (64, 16), block 128.
__global__ __launch_bounds__(128) void outproj_kernel(const float* __restrict__ attn,
                                                      const float* __restrict__ Wo,
                                                      float* __restrict__ out) {
  const int rt = blockIdx.x;
  const int cb = blockIdx.y;
  const int lane = threadIdx.x & 31;
  const int wave = threadIdx.x >> 5;
  const int col0 = cb * 128 + wave * 32;
  const int r0 = rt * 16;
  const int NHH = NHEADS * HDIM;  // 2048

  __shared__ __bf16 As[16 * 32];
  v8f acc0 = {};
  v8f acc1 = {};

  for (int kk = 0; kk < NHH; kk += 32) {
    __syncthreads();
    for (int i = threadIdx.x; i < 512; i += 128) {
      const int mr = i >> 5, kc = i & 31;
      As[i] = (__bf16)attn[(size_t)(r0 + mr) * NHH + kk + kc];
    }
    __syncthreads();
    v16bf a = load_a_lds_bf16(As, 32, lane);
    v16bf b0 = load_b_f32(Wo + (size_t)kk * DDIM + col0, DDIM, 1, lane);
    v16bf b1 = load_b_f32(Wo + (size_t)kk * DDIM + col0 + 16, DDIM, 1, lane);
    acc0 = wmma_bf16(a, b0, acc0);
    acc1 = wmma_bf16(a, b1, acc1);
  }

  const int n = lane & 15, mb = (lane >> 4) << 3;
#pragma unroll
  for (int r = 0; r < 8; ++r) {
    out[(size_t)(r0 + mb + r) * DDIM + col0 + n] = acc0[r];
    out[(size_t)(r0 + mb + r) * DDIM + col0 + 16 + n] = acc1[r];
  }
}

// ---------------- launcher ------------------------------------------------------
extern "C" void kernel_launch(void* const* d_in, const int* in_sizes, int n_in,
                              void* d_out, int out_size, void* d_ws, size_t ws_size,
                              hipStream_t stream) {
  const float* x_noise = (const float*)d_in[0];
  const float* target_hidden = (const float*)d_in[1];
  const int* noise_pos = (const int*)d_in[2];
  const int* ctx_pos = (const int*)d_in[3];
  const float* kin = (const float*)d_in[4];
  const float* vin = (const float*)d_in[5];
  const int* p_cache_len = (const int*)d_in[6];
  const int* p_actx = (const int*)d_in[7];
  const float* Wq = (const float*)d_in[8];
  const float* Wk = (const float*)d_in[9];
  const float* Wv = (const float*)d_in[10];
  const float* Wo = (const float*)d_in[11];
  const float* q_scale = (const float*)d_in[12];
  const float* k_scale = (const float*)d_in[13];

  float* out = (float*)d_out;
  const size_t OUT_ELEMS = (size_t)TNOISE * DDIM;                // 2,097,152
  const size_t KV_ELEMS = (size_t)NHEADS * MAXKV * HDIM;         // 8,388,608
  float* kout = out + OUT_ELEMS;
  float* vout = kout + KV_ELEMS;

  float* q_ws = (float*)d_ws;                                    // 8 MB
  float* attn_ws = q_ws + (size_t)TNOISE * NHEADS * HDIM;        // +8 MB

  // 1) seed output caches with the input caches (scatter overwrites updated rows)
  hipMemcpyAsync(kout, kin, KV_ELEMS * sizeof(float), hipMemcpyDeviceToDevice,
                 stream);
  hipMemcpyAsync(vout, vin, KV_ELEMS * sizeof(float), hipMemcpyDeviceToDevice,
                 stream);

  // 2) projections + norm/rope + cache update
  proj_kernel<<<dim3(TNEW / 16, NHEADS, 3), 128, 0, stream>>>(
      x_noise, target_hidden, noise_pos, ctx_pos, Wq, Wk, Wv, q_scale, k_scale,
      p_cache_len, p_actx, q_ws, kout, vout);

  // 3) attention over the updated caches
  attn_kernel<<<dim3(NHEADS, TNOISE / 16), 128, 0, stream>>>(
      q_ws, kout, vout, p_cache_len, p_actx, attn_ws);

  // 4) output projection
  outproj_kernel<<<dim3(TNOISE / 16, DDIM / 128), 128, 0, stream>>>(attn_ws, Wo,
                                                                    out);
}
